// Qwen3_6MoeRoutedExperts_89764816486464
// MI455X (gfx1250) — matrix operand order repaired and checked by reading, not compiled, not run on hardware
//
#include <hip/hip_runtime.h>
#include <stdint.h>

// ---------------------------------------------------------------------------
// MoE routed expert, bf16x3 split-precision WMMA (CDNA5 / gfx1250, wave32):
//   gu  = x @ W_gu^T   -> SwiGLU -> h (stored as bf16 hi/lo pair in scratch)
//   out = h @ W_down^T
// Each fp32 operand is split a = a_hi + a_lo (bf16); products accumulate
// a_hi*b_hi + a_hi*b_lo + a_lo*b_hi in the f32 WMMA accumulator
// (V_WMMA_F32_16X16X32_BF16), giving ~fp32 accuracy at bf16 matrix rate.
// Kernel 2 A-tiles are staged with the Tensor Data Mover when available.
// ---------------------------------------------------------------------------

typedef __bf16 bf16_t;
typedef bf16_t v4bf  __attribute__((ext_vector_type(4)));
typedef bf16_t v8bf  __attribute__((ext_vector_type(8)));
typedef bf16_t v16bf __attribute__((ext_vector_type(16)));
typedef float  v8f   __attribute__((ext_vector_type(8)));
typedef unsigned int u32x4 __attribute__((ext_vector_type(4)));
typedef int          i32x8 __attribute__((ext_vector_type(8)));
typedef int          i32x4 __attribute__((ext_vector_type(4)));

constexpr int D_MOE = 768;
constexpr int D_IN  = 2048;
constexpr int T_TOK = 4096;

// LDS row pitch: 32 bf16 data + 8 pad = 80 bytes (16B aligned; 20-bank row
// stride -> every half-wave b128 fragment read covers all 64 banks once).
constexpr int LDH = 40;

#if __has_builtin(__builtin_amdgcn_tensor_load_to_lds) && \
    __has_builtin(__builtin_amdgcn_s_wait_tensorcnt)
#define USE_TDM 1
#else
#define USE_TDM 0
#endif

__device__ __forceinline__ v8f wmma_bf16(v16bf a, v16bf b, v8f c) {
  // (neg_a, A, neg_b, B, c_mod, C, reuse_a, reuse_b)
  return __builtin_amdgcn_wmma_f32_16x16x32_bf16(false, a, false, b, (short)0,
                                                 c, false, false);
}

struct bfpair {
  bf16_t hi;
  bf16_t lo;
};

__device__ __forceinline__ bfpair split2(float f) {
  bfpair p;
  p.hi = (bf16_t)f;                 // RNE round to bf16
  p.lo = (bf16_t)(f - (float)p.hi); // exact residual, rounded to bf16
  return p;
}

// Stage a (rows x 32) fp32 tile as bf16 hi/lo tiles in LDS (pitch LDH).
__device__ __forceinline__ void stage_split_rows32(const float* __restrict__ g,
                                                   int g_ld,
                                                   bf16_t* __restrict__ lhi,
                                                   bf16_t* __restrict__ llo,
                                                   int rows, int tid, int nthr) {
  const int total = rows * 8;  // 8 float4 chunks per 32-float row
  for (int i = tid; i < total; i += nthr) {
    const int r = i >> 3;
    const int c = (i & 7) << 2;
    const float4 v = *reinterpret_cast<const float4*>(g + (size_t)r * g_ld + c);
    v4bf hi, lo;
    const bfpair p0 = split2(v.x);
    const bfpair p1 = split2(v.y);
    const bfpair p2 = split2(v.z);
    const bfpair p3 = split2(v.w);
    hi[0] = p0.hi; lo[0] = p0.lo;
    hi[1] = p1.hi; lo[1] = p1.lo;
    hi[2] = p2.hi; lo[2] = p2.lo;
    hi[3] = p3.hi; lo[3] = p3.lo;
    *reinterpret_cast<v4bf*>(&lhi[r * LDH + c]) = hi;
    *reinterpret_cast<v4bf*>(&llo[r * LDH + c]) = lo;
  }
}

// Plain bf16 (rows x 32) tile copy, global -> LDS (fallback path).
__device__ __forceinline__ void copy_bf16_rows32(const bf16_t* __restrict__ g,
                                                 int g_ld,
                                                 bf16_t* __restrict__ l,
                                                 int rows, int tid, int nthr) {
  const int total = rows * 4;  // 4 x 16B chunks per 64B row
  for (int i = tid; i < total; i += nthr) {
    const int r = i >> 2;
    const int c = (i & 3) << 3;  // 8 bf16 per chunk
    *reinterpret_cast<v8bf*>(&l[r * LDH + c]) =
        *reinterpret_cast<const v8bf*>(&g[(size_t)r * g_ld + c]);
  }
}

// A fragment (16x32 bf16, ISA 7.12.2): lane-half lh holds K = lh*8+0..7 and
// K = 16+lh*8+0..7 of row `row`.
__device__ __forceinline__ v16bf load_a_frag(const bf16_t* tile, int row, int lh) {
  const bf16_t* p = tile + row * LDH;
  const v8bf c0 = *reinterpret_cast<const v8bf*>(p + lh * 8);
  const v8bf c1 = *reinterpret_cast<const v8bf*>(p + 16 + lh * 8);
  return __builtin_shufflevector(c0, c1, 0, 1, 2, 3, 4, 5, 6, 7, 8, 9, 10, 11,
                                 12, 13, 14, 15);
}

// B fragment (32x16 bf16): lane-half lh holds K = lh*8+0..7 and
// K = 16+lh*8+0..7 of column `col` (LDS tile stored [N][K]).
__device__ __forceinline__ v16bf load_b_frag(const bf16_t* tile, int col, int lh) {
  const bf16_t* p = tile + col * LDH + lh * 8;
  const v8bf c0 = *reinterpret_cast<const v8bf*>(p);
  const v8bf c1 = *reinterpret_cast<const v8bf*>(p + 16);
  return __builtin_shufflevector(c0, c1, 0, 1, 2, 3, 4, 5, 6, 7, 8, 9, 10, 11,
                                 12, 13, 14, 15);
}

__device__ __forceinline__ float silu(float v) { return v / (1.0f + __expf(-v)); }

#if USE_TDM
// 2D Tensor-DMA load of a (rows x row_units*4B) tile into LDS with the row
// pitch padded to LDH bf16 (pad_interval = 16 DWORDs, pad_amount = 4 DWORDs).
__device__ __forceinline__ void tdm_load_tile_2d(uint32_t lds_off, const void* gptr,
                                                 uint32_t row_units, uint32_t rows,
                                                 uint32_t tensor_w_units,
                                                 uint32_t tensor_h,
                                                 uint32_t stride_units) {
  const uint64_t ga = (uint64_t)(uintptr_t)gptr;
  u32x4 g0;
  g0[0] = 1u;                                   // count=1 valid descriptor
  g0[1] = lds_off;                              // lds_addr
  g0[2] = (uint32_t)ga;                         // global_addr[31:0]
  g0[3] = (uint32_t)((ga >> 32) & 0x1FFFFFFu)   // global_addr[56:32]
          | (2u << 30);                         // type = 2 (image)
  i32x8 g1;
  g1[0] = (int)((2u << 16)      // data_size = 4B units
                | (1u << 20)    // pad_enable
                | (3u << 22)    // pad_interval: every 16 DWORDs (64B)
                | (3u << 25));  // pad_amount: 4 DWORDs (16B)
  g1[1] = (int)((tensor_w_units & 0xFFFFu) << 16);            // dim0[15:0]
  g1[2] = (int)(((tensor_w_units >> 16) & 0xFFFFu)            // dim0[31:16]
                | ((tensor_h & 0xFFFFu) << 16));              // dim1[15:0]
  g1[3] = (int)(((tensor_h >> 16) & 0xFFFFu)                  // dim1[31:16]
                | ((row_units & 0xFFFFu) << 16));             // tile_dim0
  g1[4] = (int)(rows & 0xFFFFu);                              // tile_dim1 (dim2=0)
  g1[5] = (int)stride_units;                                  // dim0_stride[31:0]
  g1[6] = 0;                                                  // stride hi bits
  g1[7] = 0;
  const i32x4 z4 = {};
  const i32x8 z8 = {};
  // amdgpu-toolchain 6-arg form: (g0, g1, g2, g3, extra, cpol)
  __builtin_amdgcn_tensor_load_to_lds(g0, g1, z4, z4, z8, 0);
}
#endif

// ---------------------------------------------------------------------------
// Kernel 1: fused gate/up projection + SwiGLU -> h (bf16 hi/lo) in scratch.
// Block tile 128 tokens x 64 moe-features (both halves), 8 waves, wave tile
// 32x32 per half; K tile = 32; 24 bf16 WMMAs / wave / K-tile.
// ---------------------------------------------------------------------------
__global__ __launch_bounds__(256)
void moe_gateup_silu_kernel(const float* __restrict__ x,
                            const float* __restrict__ gup, // (E, 2*D_MOE, D_IN)
                            const int* __restrict__ eidx,
                            bf16_t* __restrict__ h_hi,     // (T, D_MOE)
                            bf16_t* __restrict__ h_lo) {
  __shared__ __align__(16) bf16_t Ah[128 * LDH], Al[128 * LDH];
  __shared__ __align__(16) bf16_t Bgh[64 * LDH], Bgl[64 * LDH];
  __shared__ __align__(16) bf16_t Buh[64 * LDH], Bul[64 * LDH];

  const int tid  = threadIdx.x;
  const int lane = tid & 31;
  const int wid  = tid >> 5;
  const int wm   = wid & 3;
  const int wn   = wid >> 2;
  const int lm   = lane & 15;
  const int lh   = lane >> 4;

  const int t0 = blockIdx.x * 128;
  const int n0 = blockIdx.y * 64;

  const float* W  = gup + (size_t)eidx[0] * (2 * D_MOE) * D_IN;
  const float* Wg = W + (size_t)n0 * D_IN;
  const float* Wu = W + (size_t)(D_MOE + n0) * D_IN;
  const float* Xb = x + (size_t)t0 * D_IN;

  v8f ag[2][2] = {};
  v8f au[2][2] = {};

  constexpr int KT = D_IN / 32;
  for (int kt = 0; kt < KT; ++kt) {
    const int k0 = kt * 32;
    stage_split_rows32(Xb + k0, D_IN, Ah, Al, 128, tid, 256);
    stage_split_rows32(Wg + k0, D_IN, Bgh, Bgl, 64, tid, 256);
    stage_split_rows32(Wu + k0, D_IN, Buh, Bul, 64, tid, 256);
    if (kt + 1 < KT) {
      __builtin_prefetch(Xb + (size_t)(tid >> 1) * D_IN + k0 + 32, 0, 0);
      __builtin_prefetch(Wg + (size_t)(tid & 63) * D_IN + k0 + 32, 0, 0);
    }
    __syncthreads();

    v16bf ah[2], al2[2];
#pragma unroll
    for (int mi = 0; mi < 2; ++mi) {
      const int row = wm * 32 + mi * 16 + lm;
      ah[mi]  = load_a_frag(Ah, row, lh);
      al2[mi] = load_a_frag(Al, row, lh);
    }
    v16bf bgh[2], bgl[2], buh[2], bul[2];
#pragma unroll
    for (int ni = 0; ni < 2; ++ni) {
      const int col = wn * 32 + ni * 16 + lm;
      bgh[ni] = load_b_frag(Bgh, col, lh);
      bgl[ni] = load_b_frag(Bgl, col, lh);
      buh[ni] = load_b_frag(Buh, col, lh);
      bul[ni] = load_b_frag(Bul, col, lh);
    }
#pragma unroll
    for (int mi = 0; mi < 2; ++mi)
#pragma unroll
      for (int ni = 0; ni < 2; ++ni) {
        ag[mi][ni] = wmma_bf16(ah[mi],  bgh[ni], ag[mi][ni]);
        ag[mi][ni] = wmma_bf16(ah[mi],  bgl[ni], ag[mi][ni]);
        ag[mi][ni] = wmma_bf16(al2[mi], bgh[ni], ag[mi][ni]);
        au[mi][ni] = wmma_bf16(ah[mi],  buh[ni], au[mi][ni]);
        au[mi][ni] = wmma_bf16(ah[mi],  bul[ni], au[mi][ni]);
        au[mi][ni] = wmma_bf16(al2[mi], buh[ni], au[mi][ni]);
      }
    __syncthreads();
  }

  // Epilogue: h = silu(gate)*up, split to bf16 hi/lo, scatter per C/D layout.
#pragma unroll
  for (int mi = 0; mi < 2; ++mi) {
#pragma unroll
    for (int ni = 0; ni < 2; ++ni) {
      const int row0 = t0 + wm * 32 + mi * 16 + lh * 8;
      const int col  = n0 + wn * 32 + ni * 16 + lm;
      const v8f g = ag[mi][ni];
      const v8f u = au[mi][ni];
#pragma unroll
      for (int r = 0; r < 8; ++r) {
        const float hv = silu(g[r]) * u[r];
        const bfpair p = split2(hv);
        const size_t idx = (size_t)(row0 + r) * D_MOE + col;
        h_hi[idx] = p.hi;
        h_lo[idx] = p.lo;
      }
    }
  }
}

// ---------------------------------------------------------------------------
// Kernel 2: out = h @ W_down^T.  Block tile 128x128, K = 768 (24 K-tiles).
// h hi/lo tiles staged by the Tensor Data Mover (wave 0 issues, tensorcnt
// fence before the workgroup barrier); W_down split in-flight.
// ---------------------------------------------------------------------------
__global__ __launch_bounds__(256)
void moe_down_kernel(const bf16_t* __restrict__ h_hi,  // (T, D_MOE)
                     const bf16_t* __restrict__ h_lo,
                     const float* __restrict__ down,   // (E, D_IN, D_MOE)
                     const int* __restrict__ eidx,
                     float* __restrict__ out) {        // (T, D_IN)
  __shared__ __align__(16) bf16_t Ah[128 * LDH], Al[128 * LDH];
  __shared__ __align__(16) bf16_t Bh[128 * LDH], Bl[128 * LDH];

  const int tid  = threadIdx.x;
  const int lane = tid & 31;
  const int wid  = tid >> 5;
  const int wm   = wid & 3;
  const int wn   = wid >> 2;
  const int lm   = lane & 15;
  const int lh   = lane >> 4;

  const int t0 = blockIdx.x * 128;
  const int n0 = blockIdx.y * 128;

  const float* W   = down + (size_t)eidx[0] * D_IN * D_MOE + (size_t)n0 * D_MOE;
  const bf16_t* Hh = h_hi + (size_t)t0 * D_MOE;
  const bf16_t* Hl = h_lo + (size_t)t0 * D_MOE;

  v8f acc[2][4] = {};

  constexpr int KT = D_MOE / 32;
  for (int kt = 0; kt < KT; ++kt) {
    const int k0 = kt * 32;
#if USE_TDM
    if (wid == 0) {
      tdm_load_tile_2d((uint32_t)(uintptr_t)(void*)Ah, Hh + k0,
                       /*row_units=*/16, /*rows=*/128,
                       /*tensor_w_units=*/D_MOE / 2, /*tensor_h=*/T_TOK,
                       /*stride_units=*/D_MOE / 2);
      tdm_load_tile_2d((uint32_t)(uintptr_t)(void*)Al, Hl + k0,
                       16, 128, D_MOE / 2, T_TOK, D_MOE / 2);
    }
#else
    copy_bf16_rows32(Hh + k0, D_MOE, Ah, 128, tid, 256);
    copy_bf16_rows32(Hl + k0, D_MOE, Al, 128, tid, 256);
#endif
    stage_split_rows32(W + k0, D_MOE, Bh, Bl, 128, tid, 256);
    if (kt + 1 < KT) {
      __builtin_prefetch(W + (size_t)(tid >> 1) * D_MOE + k0 + 32, 0, 0);
    }
#if USE_TDM
    if (wid == 0) __builtin_amdgcn_s_wait_tensorcnt(0);
#endif
    __syncthreads();

    v16bf ah[2], al2[2];
#pragma unroll
    for (int mi = 0; mi < 2; ++mi) {
      const int row = wm * 32 + mi * 16 + lm;
      ah[mi]  = load_a_frag(Ah, row, lh);
      al2[mi] = load_a_frag(Al, row, lh);
    }
    v16bf bh[4], bl[4];
#pragma unroll
    for (int ni = 0; ni < 4; ++ni) {
      const int col = wn * 64 + ni * 16 + lm;
      bh[ni] = load_b_frag(Bh, col, lh);
      bl[ni] = load_b_frag(Bl, col, lh);
    }
#pragma unroll
    for (int mi = 0; mi < 2; ++mi)
#pragma unroll
      for (int ni = 0; ni < 4; ++ni) {
        acc[mi][ni] = wmma_bf16(ah[mi],  bh[ni], acc[mi][ni]);
        acc[mi][ni] = wmma_bf16(ah[mi],  bl[ni], acc[mi][ni]);
        acc[mi][ni] = wmma_bf16(al2[mi], bh[ni], acc[mi][ni]);
      }
    __syncthreads();
  }

#pragma unroll
  for (int mi = 0; mi < 2; ++mi) {
#pragma unroll
    for (int ni = 0; ni < 4; ++ni) {
      const int row0 = t0 + wm * 32 + mi * 16 + lh * 8;
      const int col  = n0 + wn * 64 + ni * 16 + lm;
      const v8f c = acc[mi][ni];
#pragma unroll
      for (int r = 0; r < 8; ++r) {
        out[(size_t)(row0 + r) * D_IN + col] = c[r];
      }
    }
  }
}

extern "C" void kernel_launch(void* const* d_in, const int* in_sizes, int n_in,
                              void* d_out, int out_size, void* d_ws, size_t ws_size,
                              hipStream_t stream) {
  const float* x    = (const float*)d_in[0];
  const float* gup  = (const float*)d_in[1];
  const float* down = (const float*)d_in[2];
  const int*   eidx = (const int*)d_in[3];
  float* out = (float*)d_out;

  bf16_t* h_hi = (bf16_t*)d_ws;                    // T*D_MOE bf16
  bf16_t* h_lo = h_hi + (size_t)T_TOK * D_MOE;     // T*D_MOE bf16 (12.6MB total)

  (void)in_sizes; (void)n_in; (void)out_size; (void)ws_size;

  dim3 blk(256);
  dim3 g1(T_TOK / 128, D_MOE / 64);   // 32 x 12
  dim3 g2(T_TOK / 128, D_IN / 128);   // 32 x 16
  moe_gateup_silu_kernel<<<g1, blk, 0, stream>>>(x, gup, eidx, h_hi, h_lo);
  moe_down_kernel<<<g2, blk, 0, stream>>>(h_hi, h_lo, down, eidx, out);
}
